// Im2LatexModel_76252849373646
// MI455X (gfx1250) — matrix operand order repaired and checked by reading, not compiled, not run on hardware
//
#include <hip/hip_runtime.h>
#include <hip/hip_bf16.h>
#include <math.h>

typedef _Float16 h16;
typedef __attribute__((ext_vector_type(16))) _Float16 v16h;
typedef __attribute__((ext_vector_type(8)))  _Float16 v8h;
typedef __attribute__((ext_vector_type(8)))  float    v8f;

union AFrag { v16h v; h16 e[16]; };
union CFrag { v8f  v; float e[8]; };

// CDNA5 wave32 16-bit A-matrix (16x32) element->K mapping:
// lanes 0-15 : halves {K+0..7, K+16..23};  lanes 16-31: {K+8..15, K+24..31}
__device__ __forceinline__ int a_kmap(int lane, int j) {
    int k = (j < 8) ? j : (j + 8);
    return (lane & 16) ? (k + 8) : k;
}
// B-matrix (32x16): same K mapping as A (lanes 0-15: {K+0..7, K+16..23}; lanes 16-31: {K+8..15, K+24..31})
__device__ __forceinline__ int b_kmap(int lane, int j) {
    return a_kmap(lane, j);
}

// ---------------------------------------------------------------------------
// elementwise helpers
// ---------------------------------------------------------------------------
__global__ void k_cvt_f16(const float* __restrict__ src, h16* __restrict__ dst, int n) {
    for (int i = blockIdx.x * blockDim.x + threadIdx.x; i < n; i += gridDim.x * blockDim.x)
        dst[i] = (h16)src[i];
}

// pack f32 [K,N] weight into WMMA-B fragment order
__global__ void k_pack_b(const float* __restrict__ src, h16* __restrict__ dst, int K, int N) {
    const int Nt = N >> 4;
    const int Kc = (K + 31) >> 5;
    const int total = Kc * Nt * 512;
    for (int i = blockIdx.x * blockDim.x + threadIdx.x; i < total; i += gridDim.x * blockDim.x) {
        int j = i & 15;
        int lane = (i >> 4) & 31;
        int t = i >> 9;                 // kc*Nt + nt
        int nt = t % Nt;
        int kc = t / Nt;
        int k = (kc << 5) + b_kmap(lane, j);
        int n = (nt << 4) + (lane & 15);
        dst[i] = (k < K) ? (h16)src[k * N + n] : (h16)0.f;
    }
}

__global__ void k_prep_bss(const float* __restrict__ b, const float* __restrict__ g,
                           const float* __restrict__ beta, const float* __restrict__ mean,
                           const float* __restrict__ var, float* __restrict__ bss, int C) {
    for (int c = blockIdx.x * blockDim.x + threadIdx.x; c < C; c += gridDim.x * blockDim.x) {
        float sc = g[c] * rsqrtf(var[c] + 1e-3f);
        bss[c] = b[c];
        bss[C + c] = sc;
        bss[2 * C + c] = beta[c] - mean[c] * sc;
    }
}

// ---------------------------------------------------------------------------
// conv (generic, scalar A gather) -- used only for conv1 (Cin=1, K=9)
// ---------------------------------------------------------------------------
__global__ void conv_generic_wmma(const h16* __restrict__ in, const h16* __restrict__ wpack,
                                  const float* __restrict__ bss, h16* __restrict__ out,
                                  int H, int W, int Cin, int Cout) {
    const int lane   = threadIdx.x & 31;
    const int wave   = (blockIdx.x * blockDim.x + threadIdx.x) >> 5;
    const int nwaves = (gridDim.x * blockDim.x) >> 5;
    const int HW = H * W;
    const int M  = 8 * HW;
    const int K  = 9 * Cin;
    const int Mt = M >> 4;
    const int Nt = Cout >> 4;
    const int ntiles = Mt * Nt;

    for (int tile = wave; tile < ntiles; tile += nwaves) {
        const int m0 = (tile / Nt) << 4;
        const int nt = tile - (tile / Nt) * Nt;
        const int row = m0 + (lane & 15);
        int bb = row / HW; int rem = row - bb * HW;
        int y = rem / W;   int x = rem - y * W;

        CFrag acc;
        #pragma unroll
        for (int r = 0; r < 8; ++r) acc.e[r] = 0.f;

        const int Kc = (K + 31) >> 5;
        for (int kc = 0; kc < Kc; ++kc) {
            AFrag a, b;
            const h16* bp = wpack + (((size_t)kc * Nt + nt) * 32 + lane) * 16;
            *(v8h*)&b.e[0] = *(const v8h*)(bp);
            *(v8h*)&b.e[8] = *(const v8h*)(bp + 8);
            for (int j = 0; j < 16; ++j) {
                int ka = (kc << 5) + a_kmap(lane, j);
                h16 av = (h16)0.f;
                if (ka < K) {
                    int ky = ka / (3 * Cin);
                    int rr = ka - ky * 3 * Cin;
                    int kx = rr / Cin;
                    int ci = rr - kx * Cin;
                    int iy = y + ky - 1, ix = x + kx - 1;
                    if ((unsigned)iy < (unsigned)H && (unsigned)ix < (unsigned)W)
                        av = in[((bb * H + iy) * W + ix) * Cin + ci];
                }
                a.e[j] = av;
            }
            acc.v = __builtin_amdgcn_wmma_f32_16x16x32_f16(
                false, a.v, false, b.v, (short)0, acc.v, false, false);
        }

        const int n = (nt << 4) + (lane & 15);
        const float bias  = bss[n];
        const float scale = bss[Cout + n];
        const float shift = bss[2 * Cout + n];
        #pragma unroll
        for (int r = 0; r < 8; ++r) {
            int m = m0 + r + ((lane >> 4) << 3);
            float v = fmaxf(acc.e[r] + bias, 0.f);
            out[(size_t)m * Cout + n] = (h16)(v * scale + shift);
        }
    }
}

// ---------------------------------------------------------------------------
// conv main path: block-cooperative, async Global->LDS double-buffered B panel
// ---------------------------------------------------------------------------
__global__ void __launch_bounds__(256)
conv_lds_wmma(const h16* __restrict__ in, const h16* __restrict__ wpack,
              const float* __restrict__ bss, h16* __restrict__ out,
              int H, int W, int Cin, int Cout, int cshift) {
    __shared__ h16 sB[2][8][2048];     // 64 KB: [buf][kc-local][4 tiles * 512]
    const int tid  = threadIdx.x;
    const int lane = tid & 31;
    const int wave = tid >> 5;
    const int HW = H * W;
    const int M  = 8 * HW;
    const int Mt = M >> 4;
    const int MB = Mt >> 3;            // blocks of 8 m-tiles (128 rows)
    const int Nt = Cout >> 4;
    const int NG = Nt >> 2;
    const int Kc = (9 * Cin) >> 5;
    const int nsuper = MB * NG;
    const int aoff = (lane & 16) ? 8 : 0;
    const unsigned ldsBase = (unsigned)(unsigned long long)&sB[0][0][0];
    const unsigned long long gbase = (unsigned long long)wpack;

    for (int sp = blockIdx.x; sp < nsuper; sp += gridDim.x) {
        const int mb = sp / NG;
        const int ng = sp - mb * NG;
        const int nt0 = ng << 2;
        const int m0 = (mb * 8 + wave) << 4;
        const int row = m0 + (lane & 15);
        int bb = row / HW; int rem = row - bb * HW;
        int y = rem / W;   int x = rem - y * W;

        CFrag acc[4];
        #pragma unroll
        for (int g = 0; g < 4; ++g)
            #pragma unroll
            for (int r = 0; r < 8; ++r) acc[g].e[r] = 0.f;

        const int nch = (Kc + 7) >> 3;

        {   // stage chunk 0 into buffer 0
            int kcn = Kc < 8 ? Kc : 8;
            for (int u = tid; u < kcn * 256; u += 256) {
                int kcl = u >> 8, seg = u & 255;
                unsigned lds = ldsBase + (unsigned)((kcl * 2048 + seg * 8) * 2);
                unsigned gof = (unsigned)((((size_t)kcl * Nt + nt0) * 512 + (size_t)seg * 8) * 2);
                asm volatile("global_load_async_to_lds_b128 %0, %1, %2"
                             :: "v"(lds), "v"(gof), "s"(gbase) : "memory");
            }
            asm volatile("s_wait_asynccnt 0" ::: "memory");
            __syncthreads();
        }

        int p = 0;
        for (int c = 0; c < nch; ++c) {
            const int kcLo = c << 3;
            int kcn = Kc - kcLo; if (kcn > 8) kcn = 8;

            if (c + 1 < nch) {   // prefetch next chunk (overlaps with compute)
                int nLo = kcLo + 8;
                int nn = Kc - nLo; if (nn > 8) nn = 8;
                for (int u = tid; u < nn * 256; u += 256) {
                    int kcl = u >> 8, seg = u & 255;
                    unsigned lds = ldsBase +
                        (unsigned)((((p ^ 1) * 8 + kcl) * 2048 + seg * 8) * 2);
                    unsigned gof = (unsigned)((((size_t)(nLo + kcl) * Nt + nt0) * 512 +
                                               (size_t)seg * 8) * 2);
                    asm volatile("global_load_async_to_lds_b128 %0, %1, %2"
                                 :: "v"(lds), "v"(gof), "s"(gbase) : "memory");
                }
            }

            for (int kcl = 0; kcl < kcn; ++kcl) {
                const int kc = kcLo + kcl;
                const int tap = kc >> cshift;                 // uniform
                const int cc = (kc - (tap << cshift)) << 5;
                const int ky = tap / 3, kx = tap - ky * 3;
                const int iy = y + ky - 1, ix = x + kx - 1;
                AFrag a;
                if ((unsigned)iy < (unsigned)H && (unsigned)ix < (unsigned)W) {
                    const h16* ap = in + ((size_t)(bb * H + iy) * W + ix) * Cin + cc;
                    *(v8h*)&a.e[0] = *(const v8h*)(ap + aoff);
                    *(v8h*)&a.e[8] = *(const v8h*)(ap + 16 + aoff);
                } else {
                    #pragma unroll
                    for (int j = 0; j < 16; ++j) a.e[j] = (h16)0.f;
                }
                const h16* bp = &sB[p][kcl][lane * 16];
                AFrag bfr[4];
                #pragma unroll
                for (int g = 0; g < 4; ++g) {
                    *(v8h*)&bfr[g].e[0] = *(const v8h*)(bp + g * 512);
                    *(v8h*)&bfr[g].e[8] = *(const v8h*)(bp + g * 512 + 8);
                }
                #pragma unroll
                for (int g = 0; g < 4; ++g)
                    acc[g].v = __builtin_amdgcn_wmma_f32_16x16x32_f16(
                        false, a.v, false, bfr[g].v, (short)0, acc[g].v, false, false);
            }

            asm volatile("s_wait_asynccnt 0" ::: "memory");
            __syncthreads();
            p ^= 1;
        }

        #pragma unroll
        for (int g = 0; g < 4; ++g) {
            const int n = ((nt0 + g) << 4) + (lane & 15);
            const float bias  = bss[n];
            const float scale = bss[Cout + n];
            const float shift = bss[2 * Cout + n];
            #pragma unroll
            for (int r = 0; r < 8; ++r) {
                int m = m0 + r + ((lane >> 4) << 3);
                float v = fmaxf(acc[g].e[r] + bias, 0.f);
                out[(size_t)m * Cout + n] = (h16)(v * scale + shift);
            }
        }
    }
}

// ---------------------------------------------------------------------------
// max pool, f16
// ---------------------------------------------------------------------------
__global__ void k_maxpool(const h16* __restrict__ in, h16* __restrict__ out,
                          int H, int W, int C, int ph, int pw) {
    const int Ho = H / ph, Wo = W / pw;
    const int total = 8 * Ho * Wo * C;
    for (int i = blockIdx.x * blockDim.x + threadIdx.x; i < total; i += gridDim.x * blockDim.x) {
        int c = i % C; int r = i / C;
        int xo = r % Wo; r /= Wo;
        int yo = r % Ho; int b = r / Ho;
        float m = -1e30f;
        for (int py = 0; py < ph; ++py)
            for (int px = 0; px < pw; ++px)
                m = fmaxf(m, (float)in[((b * H + yo * ph + py) * W + xo * pw + px) * C + c]);
        out[i] = (h16)m;
    }
}

// ---------------------------------------------------------------------------
// GEMM: C[M,N] = A[M,K](f16) x Bpacked + bias, f32 out. 4 N-tiles per wave.
// ---------------------------------------------------------------------------
__global__ void gemm_wmma(const h16* __restrict__ A, const h16* __restrict__ Bp,
                          const float* __restrict__ bias, float* __restrict__ C,
                          int M, int N, int K) {
    const int lane   = threadIdx.x & 31;
    const int wave   = (blockIdx.x * blockDim.x + threadIdx.x) >> 5;
    const int nwaves = (gridDim.x * blockDim.x) >> 5;
    const int Mt = M >> 4;
    const int Nt = N >> 4;
    const int NG = Nt >> 2;
    const int Kc = K >> 5;
    const int ntasks = Mt * NG;
    const int aoff = (lane & 16) ? 8 : 0;

    for (int task = wave; task < ntasks; task += nwaves) {
        const int mt = task / NG;
        const int ng = task - mt * NG;
        const int m0 = mt << 4;
        const int nt0 = ng << 2;
        const h16* arow = A + (size_t)(m0 + (lane & 15)) * K;

        CFrag acc[4];
        #pragma unroll
        for (int g = 0; g < 4; ++g)
            #pragma unroll
            for (int r = 0; r < 8; ++r) acc[g].e[r] = 0.f;

        for (int kc = 0; kc < Kc; ++kc) {
            AFrag a;
            *(v8h*)&a.e[0] = *(const v8h*)(arow + (kc << 5) + aoff);
            *(v8h*)&a.e[8] = *(const v8h*)(arow + (kc << 5) + 16 + aoff);
            const h16* bp = Bp + (((size_t)kc * Nt + nt0) * 32 + lane) * 16;
            AFrag bfr[4];
            #pragma unroll
            for (int g = 0; g < 4; ++g) {
                *(v8h*)&bfr[g].e[0] = *(const v8h*)(bp + g * 512);
                *(v8h*)&bfr[g].e[8] = *(const v8h*)(bp + g * 512 + 8);
            }
            #pragma unroll
            for (int g = 0; g < 4; ++g)
                acc[g].v = __builtin_amdgcn_wmma_f32_16x16x32_f16(
                    false, a.v, false, bfr[g].v, (short)0, acc[g].v, false, false);
        }

        #pragma unroll
        for (int g = 0; g < 4; ++g) {
            const int n = ((nt0 + g) << 4) + (lane & 15);
            const float bv = bias ? bias[n] : 0.f;
            #pragma unroll
            for (int r = 0; r < 8; ++r) {
                int m = m0 + r + ((lane >> 4) << 3);
                C[(size_t)m * N + n] = acc[g].e[r] + bv;
            }
        }
    }
}

// permute proj output [B*Hf*Wf, D] -> row-scan order [Wf][B*Hf][D], f16
__global__ void k_perm_rows(const float* __restrict__ x, h16* __restrict__ xr) {
    const int total = 8192 * 256;
    for (int i = blockIdx.x * blockDim.x + threadIdx.x; i < total; i += gridDim.x * blockDim.x) {
        int d = i & 255; int m = i >> 8;
        int s = m >> 6;      // b*16+h
        int w = m & 63;
        xr[((w << 7) + s) * 256 + d] = (h16)x[i];
    }
}

// ---------------------------------------------------------------------------
// full bidirectional GRU: grid=2 (block 0 = fwd, block 1 = bwd), all 64 steps
// looped in-kernel.  Per step: WMMA h@U (M=128,N=384,K=128), then gate math.
// ---------------------------------------------------------------------------
__global__ void __launch_bounds__(256)
gru_run(const float* __restrict__ gxf, const float* __restrict__ gxb,
        const h16* __restrict__ Upf, const h16* __restrict__ Upb,
        const float* __restrict__ b1f, const float* __restrict__ b1b,
        float* __restrict__ ghbuf, float* __restrict__ hstbuf,
        h16* __restrict__ h16buf, float* __restrict__ fwdout,
        float* __restrict__ bwdout) {
    const int dir  = blockIdx.x;
    const float* gx = dir ? gxb : gxf;
    const h16*  Up  = dir ? Upb : Upf;
    const float* b1 = dir ? b1b : b1f;
    float* outb   = dir ? bwdout : fwdout;
    float* gh     = ghbuf + (size_t)dir * 128 * 384;
    float* hstate = hstbuf + (size_t)dir * 16384;
    h16*  h16s    = h16buf + (size_t)dir * 16384;

    const int tid  = threadIdx.x;
    const int lane = tid & 31;
    const int wave = tid >> 5;
    const int aoff = (lane & 16) ? 8 : 0;
    const int Nt = 24;

    for (int i = tid; i < 16384; i += 256) { hstate[i] = 0.f; h16s[i] = (h16)0.f; }
    __threadfence();
    __syncthreads();

    for (int step = 0; step < 64; ++step) {
        const int t = dir ? (63 - step) : step;

        for (int task = wave; task < 8 * 6; task += 8) {
            const int mt = task / 6;
            const int ng = task - mt * 6;
            const int m0 = mt << 4;
            const int nt0 = ng << 2;
            const h16* arow = h16s + (size_t)(m0 + (lane & 15)) * 128;

            CFrag acc[4];
            #pragma unroll
            for (int g = 0; g < 4; ++g)
                #pragma unroll
                for (int r = 0; r < 8; ++r) acc[g].e[r] = 0.f;

            #pragma unroll
            for (int kc = 0; kc < 4; ++kc) {
                AFrag a;
                *(v8h*)&a.e[0] = *(const v8h*)(arow + (kc << 5) + aoff);
                *(v8h*)&a.e[8] = *(const v8h*)(arow + (kc << 5) + 16 + aoff);
                const h16* bp = Up + (((size_t)kc * Nt + nt0) * 32 + lane) * 16;
                AFrag bfr[4];
                #pragma unroll
                for (int g = 0; g < 4; ++g) {
                    *(v8h*)&bfr[g].e[0] = *(const v8h*)(bp + g * 512);
                    *(v8h*)&bfr[g].e[8] = *(const v8h*)(bp + g * 512 + 8);
                }
                #pragma unroll
                for (int g = 0; g < 4; ++g)
                    acc[g].v = __builtin_amdgcn_wmma_f32_16x16x32_f16(
                        false, a.v, false, bfr[g].v, (short)0, acc[g].v, false, false);
            }

            #pragma unroll
            for (int g = 0; g < 4; ++g) {
                const int n = ((nt0 + g) << 4) + (lane & 15);
                #pragma unroll
                for (int r = 0; r < 8; ++r) {
                    int m = m0 + r + ((lane >> 4) << 3);
                    gh[m * 384 + n] = acc[g].e[r] + b1[n];
                }
            }
        }
        __threadfence();
        __syncthreads();

        const float* gxt = gx + t * 128 * 384;
        for (int i = tid; i < 128 * 128; i += 256) {
            int s = i >> 7, d = i & 127;
            float z = 1.f / (1.f + __expf(-(gxt[s * 384 + d] + gh[s * 384 + d])));
            float r = 1.f / (1.f + __expf(-(gxt[s * 384 + 128 + d] + gh[s * 384 + 128 + d])));
            float cand = tanhf(gxt[s * 384 + 256 + d] + r * gh[s * 384 + 256 + d]);
            float hn = z * hstate[i] + (1.f - z) * cand;
            hstate[i] = hn;
            h16s[i] = (h16)hn;
            outb[t * 128 * 128 + i] = hn;
        }
        __threadfence();
        __syncthreads();
    }
}

__global__ void k_build_enc(const float* __restrict__ fwd, const float* __restrict__ bwd,
                            float* __restrict__ enc, h16* __restrict__ enc16) {
    const int total = 8192 * 256;
    for (int i = blockIdx.x * blockDim.x + threadIdx.x; i < total; i += gridDim.x * blockDim.x) {
        int d = i & 255; int m = i >> 8;
        int s = m >> 6; int w = m & 63;
        float v = (d < 128) ? fwd[((w << 7) + s) * 128 + d]
                            : bwd[((w << 7) + s) * 128 + (d - 128)];
        enc[i] = v;
        enc16[i] = (h16)v;
    }
}

// ---------------------------------------------------------------------------
// fused decoder: grid=8 (one block per batch row), all 64 timesteps in-kernel.
// Everything per-batch lives in LDS; logits written straight to d_out.
// ---------------------------------------------------------------------------
__global__ void __launch_bounds__(256)
dec_all(const float* __restrict__ W1f, const float* __restrict__ enc,
        const float* __restrict__ embed, const int* __restrict__ tgt,
        const float* __restrict__ W2w, const float* __restrict__ W2b,
        const float* __restrict__ Vw, const float* __restrict__ Vb,
        const float* __restrict__ dgW, const float* __restrict__ dgU,
        const float* __restrict__ dgb,
        const float* __restrict__ f1w, const float* __restrict__ f1b,
        const float* __restrict__ f2w, const float* __restrict__ f2b,
        float* __restrict__ out) {
    __shared__ float qs[256];
    __shared__ float sc[1024];
    __shared__ float red[256];
    __shared__ float ctx[256];
    __shared__ float hid[256];
    __shared__ float gin[384];
    __shared__ float hnn[256];
    __shared__ float pre1[256];
    const int b = blockIdx.x, tid = threadIdx.x;

    hid[tid] = 0.f;
    __syncthreads();

    for (int t = 0; t < 64; ++t) {
        // q = hid @ W2 + b2
        float q = W2b[tid];
        for (int k = 0; k < 256; ++k) q += hid[k] * W2w[k * 256 + tid];
        qs[tid] = q;
        __syncthreads();

        // scores + softmax over 1024 positions
        float lmax = -1e30f;
        for (int p = tid; p < 1024; p += 256) {
            const float* wrow = W1f + (b * 1024 + p) * 256;
            float s = Vb[0];
            for (int a = 0; a < 256; ++a) s += tanhf(wrow[a] + qs[a]) * Vw[a];
            sc[p] = s;
            lmax = fmaxf(lmax, s);
        }
        red[tid] = lmax; __syncthreads();
        for (int o = 128; o; o >>= 1) {
            if (tid < o) red[tid] = fmaxf(red[tid], red[tid + o]);
            __syncthreads();
        }
        const float mx = red[0]; __syncthreads();
        float lsum = 0.f;
        for (int p = tid; p < 1024; p += 256) {
            float e = __expf(sc[p] - mx);
            sc[p] = e; lsum += e;
        }
        red[tid] = lsum; __syncthreads();
        for (int o = 128; o; o >>= 1) {
            if (tid < o) red[tid] += red[tid + o];
            __syncthreads();
        }
        const float inv = 1.f / red[0]; __syncthreads();

        // context
        float c = 0.f;
        for (int p = 0; p < 1024; ++p) c += sc[p] * enc[(b * 1024 + p) * 256 + tid];
        ctx[tid] = c * inv;
        __syncthreads();

        // gin = [embed(tok), ctx]
        const int tok = tgt[b * 64 + t];
        for (int k = tid; k < 384; k += 256)
            gin[k] = (k < 128) ? embed[tok * 128 + k] : ctx[k - 128];
        __syncthreads();

        // GRU cell (d = tid)
        {
            const int d = tid;
            float xz = dgb[d], xr = dgb[256 + d], xh = dgb[512 + d];
            for (int k = 0; k < 384; ++k) {
                float g = gin[k];
                xz += g * dgW[k * 768 + d];
                xr += g * dgW[k * 768 + 256 + d];
                xh += g * dgW[k * 768 + 512 + d];
            }
            float hz = dgb[768 + d], hr = dgb[768 + 256 + d], hh = dgb[768 + 512 + d];
            for (int k = 0; k < 256; ++k) {
                float h = hid[k];
                hz += h * dgU[k * 768 + d];
                hr += h * dgU[k * 768 + 256 + d];
                hh += h * dgU[k * 768 + 512 + d];
            }
            float z = 1.f / (1.f + __expf(-(xz + hz)));
            float r = 1.f / (1.f + __expf(-(xr + hr)));
            float cand = tanhf(xh + r * hh);
            hnn[tid] = z * hid[d] + (1.f - z) * cand;
        }
        __syncthreads();
        hid[tid] = hnn[tid];
        __syncthreads();

        // fc1: tanh([hid, ctx] @ f1w + b)
        {
            float s = f1b[tid];
            for (int k = 0; k < 256; ++k) s += hid[k] * f1w[k * 256 + tid];
            for (int k = 0; k < 256; ++k) s += ctx[k] * f1w[(256 + k) * 256 + tid];
            pre1[tid] = tanhf(s);
        }
        __syncthreads();

        // fc2 -> logits
        for (int o = tid; o < 1000; o += 256) {
            float s = f2b[o];
            for (int k = 0; k < 256; ++k) s += pre1[k] * f2w[k * 1000 + o];
            out[((size_t)b * 64 + t) * 1000 + o] = s;
        }
        __syncthreads();
    }
}

// ---------------------------------------------------------------------------
// host launch
// ---------------------------------------------------------------------------
extern "C" void kernel_launch(void* const* d_in, const int* in_sizes, int n_in,
                              void* d_out, int out_size, void* d_ws, size_t ws_size,
                              hipStream_t stream) {
    (void)in_sizes; (void)n_in; (void)out_size; (void)ws_size;

    const float* images = (const float*)d_in[0];
    const int*   tgt    = (const int*)d_in[1];
    const float *cw[6], *cb[6], *cg[6], *cbe[6], *cm[6], *cv[6];
    for (int l = 0; l < 6; ++l) {
        int base = 2 + l * 6;
        cw[l]  = (const float*)d_in[base + 0];
        cb[l]  = (const float*)d_in[base + 1];
        cg[l]  = (const float*)d_in[base + 2];
        cbe[l] = (const float*)d_in[base + 3];
        cm[l]  = (const float*)d_in[base + 4];
        cv[l]  = (const float*)d_in[base + 5];
    }
    const float* projW = (const float*)d_in[38];
    const float* projB = (const float*)d_in[39];
    const float* rfW = (const float*)d_in[40];
    const float* rfU = (const float*)d_in[41];
    const float* rfb = (const float*)d_in[42];
    const float* rbW = (const float*)d_in[43];
    const float* rbU = (const float*)d_in[44];
    const float* rbb = (const float*)d_in[45];
    const float* embed = (const float*)d_in[46];
    const float* aW1w = (const float*)d_in[47];
    const float* aW1b = (const float*)d_in[48];
    const float* aW2w = (const float*)d_in[49];
    const float* aW2b = (const float*)d_in[50];
    const float* aVw  = (const float*)d_in[51];
    const float* aVb  = (const float*)d_in[52];
    const float* dgW  = (const float*)d_in[53];
    const float* dgU  = (const float*)d_in[54];
    const float* dgb  = (const float*)d_in[55];
    const float* f1w  = (const float*)d_in[56];
    const float* f1b  = (const float*)d_in[57];
    const float* f2w  = (const float*)d_in[58];
    const float* f2b  = (const float*)d_in[59];

    char* base = (char*)d_ws;
    size_t off = 0;
    auto carve = [&](size_t bytes) -> void* {
        void* p = base + off;
        off = (off + bytes + 255) & ~(size_t)255;
        return p;
    };
    auto packsz = [](int K, int N) -> size_t {      // halves
        return (size_t)((K + 31) / 32) * 32 * N;
    };

    const int ci[6] = {1, 64, 128, 256, 256, 256};
    const int co[6] = {64, 128, 256, 256, 256, 256};
    const int csh[6] = {0, 1, 2, 3, 3, 3};

    h16*   a16  = (h16*)carve((size_t)524288 * 2);
    h16*   bufA = (h16*)carve((size_t)33554432 * 2);
    h16*   bufB = (h16*)carve((size_t)33554432 * 2);
    h16*   wfP[6]; float* bssp[6];
    for (int l = 0; l < 6; ++l) {
        wfP[l]  = (h16*)carve(packsz(9 * ci[l], co[l]) * 2);
        bssp[l] = (float*)carve((size_t)3 * co[l] * 4);
    }
    h16* projP = (h16*)carve(packsz(256, 256) * 2);
    h16* WfP   = (h16*)carve(packsz(256, 384) * 2);
    h16* WbP   = (h16*)carve(packsz(256, 384) * 2);
    h16* UfP   = (h16*)carve(packsz(128, 384) * 2);
    h16* UbP   = (h16*)carve(packsz(128, 384) * 2);
    h16* aW1P  = (h16*)carve(packsz(256, 256) * 2);
    float* xf32 = (float*)carve((size_t)8192 * 256 * 4);
    h16*  xr16  = (h16*)carve((size_t)8192 * 256 * 2);
    float* gxf  = (float*)carve((size_t)8192 * 384 * 4);
    float* gxb  = (float*)carve((size_t)8192 * 384 * 4);
    float* gh   = (float*)carve((size_t)2 * 128 * 384 * 4);
    float* hst  = (float*)carve((size_t)2 * 16384 * 4);
    h16*  hst16 = (h16*)carve((size_t)2 * 16384 * 2);
    float* fwdout = (float*)carve((size_t)64 * 16384 * 4);
    float* bwdout = (float*)carve((size_t)64 * 16384 * 4);
    float* encB  = (float*)carve((size_t)2097152 * 4);
    h16*  enc16  = (h16*)carve((size_t)2097152 * 2);
    float* W1f   = (float*)carve((size_t)2097152 * 4);

    // ---- weight prep (pack all B matrices into WMMA fragment order) ----
    for (int l = 0; l < 6; ++l) {
        int np = (int)packsz(9 * ci[l], co[l]);
        k_pack_b<<<(np + 255) / 256, 256, 0, stream>>>(cw[l], wfP[l], 9 * ci[l], co[l]);
        k_prep_bss<<<1, 256, 0, stream>>>(cb[l], cg[l], cbe[l], cm[l], cv[l], bssp[l], co[l]);
    }
    k_pack_b<<<256, 256, 0, stream>>>(projW, projP, 256, 256);
    k_pack_b<<<384, 256, 0, stream>>>(rfW, WfP, 256, 384);
    k_pack_b<<<384, 256, 0, stream>>>(rbW, WbP, 256, 384);
    k_pack_b<<<192, 256, 0, stream>>>(rfU, UfP, 128, 384);
    k_pack_b<<<192, 256, 0, stream>>>(rbU, UbP, 128, 384);
    k_pack_b<<<256, 256, 0, stream>>>(aW1w, aW1P, 256, 256);
    k_cvt_f16<<<2048, 256, 0, stream>>>(images, a16, 524288);

    // ---- CNN encoder ----
    conv_generic_wmma<<<1024, 128, 0, stream>>>(a16, wfP[0], bssp[0], bufA, 128, 512, 1, 64);
    k_maxpool<<<8192, 256, 0, stream>>>(bufA, bufB, 128, 512, 64, 2, 2);      // 64x256x64
    conv_lds_wmma<<<512, 256, 0, stream>>>(bufB, wfP[1], bssp[1], bufA, 64, 256, 64, 128, csh[1]);
    k_maxpool<<<4096, 256, 0, stream>>>(bufA, bufB, 64, 256, 128, 2, 2);      // 32x128x128
    conv_lds_wmma<<<512, 256, 0, stream>>>(bufB, wfP[2], bssp[2], bufA, 32, 128, 128, 256, csh[2]);
    conv_lds_wmma<<<512, 256, 0, stream>>>(bufA, wfP[3], bssp[3], bufB, 32, 128, 256, 256, csh[3]);
    k_maxpool<<<2048, 256, 0, stream>>>(bufB, bufA, 32, 128, 256, 2, 1);      // 16x128x256
    conv_lds_wmma<<<512, 256, 0, stream>>>(bufA, wfP[4], bssp[4], bufB, 16, 128, 256, 256, csh[4]);
    conv_lds_wmma<<<512, 256, 0, stream>>>(bufB, wfP[5], bssp[5], bufA, 16, 128, 256, 256, csh[5]);
    k_maxpool<<<1024, 256, 0, stream>>>(bufA, bufB, 16, 128, 256, 1, 2);      // 16x64x256

    // ---- row encoder ----
    gemm_wmma<<<512, 128, 0, stream>>>(bufB, projP, projB, xf32, 8192, 256, 256);
    k_perm_rows<<<4096, 256, 0, stream>>>(xf32, xr16);
    gemm_wmma<<<512, 128, 0, stream>>>(xr16, WfP, rfb, gxf, 8192, 384, 256);       // b[0] fwd
    gemm_wmma<<<512, 128, 0, stream>>>(xr16, WbP, rbb, gxb, 8192, 384, 256);       // b[0] bwd

    gru_run<<<2, 256, 0, stream>>>(gxf, gxb, UfP, UbP, rfb + 384, rbb + 384,
                                   gh, hst, hst16, fwdout, bwdout);

    k_build_enc<<<4096, 256, 0, stream>>>(fwdout, bwdout, encB, enc16);
    gemm_wmma<<<512, 128, 0, stream>>>(enc16, aW1P, aW1b, W1f, 8192, 256, 256);

    // ---- fused attention decoder (one launch, 8 blocks, 64 steps) ----
    dec_all<<<8, 256, 0, stream>>>(W1f, encB, embed, tgt, aW2w, aW2b, aVw, aVb,
                                   dgW, dgU, dgb, f1w, f1b, f2w, f2b, (float*)d_out);
}
